// Attention_27951647162775
// MI455X (gfx1250) — matrix operand order repaired and checked by reading, not compiled, not run on hardware
//
#include <hip/hip_runtime.h>
#include <hip/hip_bf16.h>

// ---------------------------------------------------------------------------
// Bahdanau attention for MI455X (gfx1250, wave32, WMMA).
// B=64, P=196, ENC=2048, DEC=2048, ATTN=1024.
//
// Dominant GEMM (12544x2048 @ 2048x1024, 52.6 GFLOP) runs on
// v_wmma_f32_16x16x32_bf16 with f32 accumulation, fused with bias +
// h_proj broadcast + tanh + V-dot score reduction so the (B,P,ATTN)
// intermediate never exists. Output-stationary 128x128 block tiles
// (grid 98x8) keep L2 traffic to ~0.8 GB. All operands pre-converted to
// bf16 so the hot loop is pure global_load_b128 + v_wmma. The K loop is
// explicitly software-pipelined two deep (double fragment buffers) so the
// next K-step's 18 loads are in flight while the current 8 WMMAs execute.
// V_b is dropped: softmax is shift-invariant, scores are not an output.
// ---------------------------------------------------------------------------

typedef __bf16  v16bf __attribute__((ext_vector_type(16)));
typedef float   v8f   __attribute__((ext_vector_type(8)));
typedef unsigned int v8u __attribute__((ext_vector_type(8)));

#define K_DIM   2048
#define N_DIM   1024
#define BATCH   64
#define SEQ_P   196
#define ROWS_BP (BATCH * SEQ_P)   // 12544 = 98 * 128
#define NKT     (K_DIM / 32)      // 64 K-steps

__device__ __forceinline__ unsigned short f2bf(float f) {
    unsigned int u = __float_as_uint(f);
    u += 0x7FFFu + ((u >> 16) & 1u);       // round-to-nearest-even
    return (unsigned short)(u >> 16);
}

// --------------------------- fp32 -> bf16 convert ---------------------------
__global__ __launch_bounds__(256, 1)
void cvt_bf16_kernel(const float* __restrict__ in,
                     unsigned short* __restrict__ out, int n4) {
    int i = blockIdx.x * blockDim.x + threadIdx.x;
    if (i >= n4) return;
    float4 v = ((const float4*)in)[i];
    unsigned int lo = (unsigned int)f2bf(v.x) | ((unsigned int)f2bf(v.y) << 16);
    unsigned int hi = (unsigned int)f2bf(v.z) | ((unsigned int)f2bf(v.w) << 16);
    ((uint2*)out)[i] = make_uint2(lo, hi);
}

__global__ void zero_kernel(float* __restrict__ p, int n) {
    int i = blockIdx.x * blockDim.x + threadIdx.x;
    if (i < n) p[i] = 0.0f;
}

// -------------------- fragment load / WMMA helpers --------------------------
// Per-lane fragment addressing (ISA 16-bit layouts, CDNA5 ISA 7.12.2):
//   A lane(m=lane&15, half=hi): K = [hi*8, +8) and [16+hi*8, +8) -> 2x b128
//   B lane(n=lane&15, half=hi): K = [hi*8, +8) and [16+hi*8, +8) -> 2x b128
template <int MT>
__device__ __forceinline__ void load_frags(const unsigned short* __restrict__ Bp,
                                           const unsigned short* __restrict__ Ap0,
                                           int ko, v8u& braw, v8u (&araw)[MT]) {
    ((uint4*)&braw)[0] = *(const uint4*)(Bp + ko);
    ((uint4*)&braw)[1] = *(const uint4*)(Bp + ko + 16);
#pragma unroll
    for (int mt = 0; mt < MT; mt++) {
        const unsigned short* ap = Ap0 + (size_t)mt * 16 * K_DIM + ko;
        ((uint4*)&araw[mt])[0] = *(const uint4*)(ap);
        ((uint4*)&araw[mt])[1] = *(const uint4*)(ap + 16);
    }
}

template <int MT>
__device__ __forceinline__ void wmma_all(const v8u& braw, const v8u (&araw)[MT],
                                         v8f (&acc)[MT]) {
    v16bf bfrag = __builtin_bit_cast(v16bf, braw);
#pragma unroll
    for (int mt = 0; mt < MT; mt++) {
        acc[mt] = __builtin_amdgcn_wmma_f32_16x16x32_bf16(
            false, __builtin_bit_cast(v16bf, araw[mt]), false, bfrag,
            (short)0, acc[mt], false, false);
    }
}

// ------------------------ WMMA GEMM (optionally fused) -----------------------
// Block tile: (MT*16) rows x 128 cols. 8 waves, wave w owns the 16-col strip
// n = n0 + w*16 + (lane&15), and all MT row-tiles (A frags shared via WGP$).
// FUSED=0: out[r*1024 + n] = acc + bias[n]                       (h_proj)
// FUSED=1: atomicAdd(out[r], sum_n tanh(acc + bias[n] + hproj[b(r)][n])*Vw[n])
template <int MT, bool FUSED>
__global__ __launch_bounds__(256, 1)
void wmma_gemm_kernel(const unsigned short* __restrict__ Abf,
                      const unsigned short* __restrict__ Bw,
                      const float* __restrict__ bias,
                      const float* __restrict__ hproj,
                      const float* __restrict__ Vw,
                      float* __restrict__ out) {
    const int tid  = threadIdx.x;
    const int wave = tid >> 5;
    const int lane = tid & 31;
    const int m_l  = lane & 15;
    const int hi   = lane >> 4;
    const int r0   = blockIdx.x * (MT * 16);
    const int n0   = blockIdx.y * 128;
    const int n    = n0 + wave * 16 + m_l;     // this lane's output column

    const unsigned short* Bp  = Bw  + (size_t)n * K_DIM + hi * 8;
    const unsigned short* Ap0 = Abf + (size_t)(r0 + m_l) * K_DIM + hi * 8;

    v8f acc[MT];
#pragma unroll
    for (int mt = 0; mt < MT; mt++)
#pragma unroll
        for (int e = 0; e < 8; e++) acc[mt][e] = 0.0f;

    // ---- software-pipelined K loop: two fragment buffers in flight ----
    v8u bA, aA[MT], bB, aB[MT];
    load_frags<MT>(Bp, Ap0, 0, bA, aA);
    for (int kt = 0; kt < NKT - 2; kt += 2) {
        load_frags<MT>(Bp, Ap0, (kt + 1) * 32, bB, aB);   // next (odd) step
        wmma_all<MT>(bA, aA, acc);                        // current even step
        load_frags<MT>(Bp, Ap0, (kt + 2) * 32, bA, aA);   // step after next
        wmma_all<MT>(bB, aB, acc);                        // odd step
    }
    load_frags<MT>(Bp, Ap0, (NKT - 1) * 32, bB, aB);
    wmma_all<MT>(bA, aA, acc);                            // step NKT-2
    wmma_all<MT>(bB, aB, acc);                            // step NKT-1

    // ------------------------------ epilogue ------------------------------
    if constexpr (!FUSED) {
        float bn = bias[n];
#pragma unroll
        for (int mt = 0; mt < MT; mt++)
#pragma unroll
            for (int i = 0; i < 8; i++) {
                int r = r0 + mt * 16 + i + 8 * hi;
                out[(size_t)r * N_DIM + n] = acc[mt][i] + bn;
            }
    } else {
        __shared__ float s_sc[MT * 16];
        const int b0 = r0 / SEQ_P;                  // block spans <= 2 batches
        const int rsplit = (b0 + 1) * SEQ_P;        // first row of batch b0+1
        const float bn = bias[n];
        const float vw = Vw[n];
        const float h0 = hproj[(size_t)b0 * N_DIM + n];
        const float h1 = hproj[(size_t)(b0 + 1 < BATCH ? b0 + 1 : b0) * N_DIM + n];

        if (tid < MT * 16) s_sc[tid] = 0.0f;
        __syncthreads();

#pragma unroll
        for (int mt = 0; mt < MT; mt++) {
            float sc[8];
#pragma unroll
            for (int i = 0; i < 8; i++) {
                int r = r0 + mt * 16 + i + 8 * hi;
                float hv = (r >= rsplit) ? h1 : h0;
                sc[i] = tanhf(acc[mt][i] + bn + hv) * vw;
            }
            // reduce over the 16 column-lanes of each half-wave
#pragma unroll
            for (int i = 0; i < 8; i++) {
                float v = sc[i];
                for (int off = 1; off < 16; off <<= 1) v += __shfl_xor(v, off, 32);
                sc[i] = v;
            }
            if (m_l == 0) {
#pragma unroll
                for (int i = 0; i < 8; i++)
                    atomicAdd(&s_sc[mt * 16 + i + 8 * hi], sc[i]);
            }
        }
        __syncthreads();
        if (tid < MT * 16) atomicAdd(&out[r0 + tid], s_sc[tid]);  // scores
    }
}

// --------------------- softmax over P + context GEMV ------------------------
__global__ __launch_bounds__(256, 1)
void softmax_ctx_kernel(const float* __restrict__ scores,
                        const float* __restrict__ enc,
                        float* __restrict__ ctx_out,
                        float* __restrict__ attn_out) {
    const int b = blockIdx.x, t = threadIdx.x;
    __shared__ float sw[SEQ_P];
    __shared__ float red[256];

    float s = (t < SEQ_P) ? scores[b * SEQ_P + t] : -3.4e38f;
    red[t] = s;
    __syncthreads();
    for (int off = 128; off > 0; off >>= 1) {
        if (t < off) red[t] = fmaxf(red[t], red[t + off]);
        __syncthreads();
    }
    float mx = red[0];
    __syncthreads();
    float e = (t < SEQ_P) ? __expf(s - mx) : 0.0f;
    red[t] = e;
    __syncthreads();
    for (int off = 128; off > 0; off >>= 1) {
        if (t < off) red[t] += red[t + off];
        __syncthreads();
    }
    float inv = 1.0f / red[0];
    float w = e * inv;
    if (t < SEQ_P) { sw[t] = w; attn_out[b * SEQ_P + t] = w; }
    __syncthreads();

    // context[b, e] = sum_p w[p] * enc[b, p, e]; thread owns e=4t..4t+3, 1024+4t..
    const float* eb = enc + (size_t)b * SEQ_P * 2048;
    float4 a0 = make_float4(0.f, 0.f, 0.f, 0.f);
    float4 a1 = make_float4(0.f, 0.f, 0.f, 0.f);
    for (int p = 0; p < SEQ_P; p++) {
        float wp = sw[p];
        const float4* row = (const float4*)(eb + (size_t)p * 2048);
        float4 x0 = row[t];
        float4 x1 = row[256 + t];
        a0.x += wp * x0.x; a0.y += wp * x0.y; a0.z += wp * x0.z; a0.w += wp * x0.w;
        a1.x += wp * x1.x; a1.y += wp * x1.y; a1.z += wp * x1.z; a1.w += wp * x1.w;
    }
    float4* o = (float4*)(ctx_out + (size_t)b * 2048);
    o[t] = a0;
    o[256 + t] = a1;
}

// ----------------------------------------------------------------------------
extern "C" void kernel_launch(void* const* d_in, const int* in_sizes, int n_in,
                              void* d_out, int out_size, void* d_ws, size_t ws_size,
                              hipStream_t stream) {
    const float* enc = (const float*)d_in[0];   // (64,196,2048)
    const float* dec = (const float*)d_in[1];   // (64,2048)
    const float* W1w = (const float*)d_in[2];   // (1024,2048)
    const float* W1b = (const float*)d_in[3];   // (1024)
    const float* W2w = (const float*)d_in[4];   // (1024,2048)
    const float* W2b = (const float*)d_in[5];   // (1024)
    const float* Vw  = (const float*)d_in[6];   // (1,1024)

    float* ctx_out  = (float*)d_out;                 // 64*2048
    float* attn_out = (float*)d_out + BATCH * 2048;  // 64*196

    char* ws = (char*)d_ws;
    unsigned short* W2bf = (unsigned short*)ws;                      // 4 MB
    unsigned short* W1bf = (unsigned short*)(ws + (4u << 20));       // 4 MB
    float* hproj          = (float*)(ws + (8u << 20));               // 256 KB
    float* scores         = (float*)(ws + (8u << 20) + (256u << 10));// 64 KB slot
    unsigned short* decbf = (unsigned short*)(ws + (8u << 20) + (320u << 10)); // 256 KB
    unsigned short* encbf = (unsigned short*)(ws + (9u << 20));      // 51.4 MB

    // bf16 conversions (RNE)
    const int w4 = (N_DIM * K_DIM) / 4;           // 524288
    cvt_bf16_kernel<<<w4 / 256, 256, 0, stream>>>(W2w, W2bf, w4);
    cvt_bf16_kernel<<<w4 / 256, 256, 0, stream>>>(W1w, W1bf, w4);
    const int d4 = (BATCH * K_DIM) / 4;           // 32768
    cvt_bf16_kernel<<<d4 / 256, 256, 0, stream>>>(dec, decbf, d4);
    const int e4 = (ROWS_BP * K_DIM) / 4;         // 6422528
    cvt_bf16_kernel<<<e4 / 256, 256, 0, stream>>>(enc, encbf, e4);

    zero_kernel<<<(ROWS_BP + 255) / 256, 256, 0, stream>>>(scores, ROWS_BP);

    // h_proj = dec @ W1^T + W1_b : 64x1024, tile 64x128 -> grid (1, 8)
    wmma_gemm_kernel<4, false><<<dim3(1, 8), 256, 0, stream>>>(
        decbf, W1bf, W1b, nullptr, nullptr, hproj);

    // scores[r] += sum_n tanh(enc@W2^T + W2_b + h_proj)[r,n] * V[n]
    // tile 128x128 -> grid (98, 8)
    wmma_gemm_kernel<8, true><<<dim3(ROWS_BP / 128, 8), 256, 0, stream>>>(
        encbf, W2bf, W2b, hproj, Vw, scores);

    softmax_ctx_kernel<<<BATCH, 256, 0, stream>>>(scores, enc, ctx_out, attn_out);
}
